// AMRSRN_74036646249218
// MI455X (gfx1250) — matrix operand order repaired and checked by reading, not compiled, not run on hardware
//
#include <hip/hip_runtime.h>
#include <hip/hip_bf16.h>

typedef _Float16 f16;
typedef __attribute__((ext_vector_type(16))) _Float16 v16h;
typedef __attribute__((ext_vector_type(8)))  _Float16 v8h;
typedef __attribute__((ext_vector_type(4)))  _Float16 v4h;
typedef __attribute__((ext_vector_type(8)))  float    v8f;
typedef int gint4 __attribute__((vector_size(4 * sizeof(int))));

#define N_POINTS_C 131072
#define PPB 128            // points per block
#define THREADS_C 256
#define ACT_STRIDE 264     // 256 + 8 f16 pad (bank-conflict mitigation)
#define KC0 6              // layer0: K=192 (164 padded), 6 chunks of 32
#define KC1 8              // layers1-5: K=256, 8 chunks of 32
#define L0_WELEMS (16*KC0*32*16)     // 49152 f16
#define L1_WELEMS (16*KC1*32*16)     // 65536 f16
#define W_F16_TOTAL (L0_WELEMS + 5*L1_WELEMS)   // 376832 f16
#define F32_TAIL (6*256 + 256 + 1)   // biases[0..5], w6, b6
#define PI_F 3.14159265358979323846f

// CDNA5 async global->LDS path (ASYNCcnt-tracked, bypasses VGPRs).
// Guarded: falls back to global_load_b128 + ds_store_b128 staging if the
// toolchain lacks the builtins.
#if defined(__gfx1250__) && __has_builtin(__builtin_amdgcn_global_load_async_to_lds_b128)
#define ASYNC_LDS 1
typedef __attribute__((address_space(1))) gint4* as1_int4p;
typedef __attribute__((address_space(3))) gint4* as3_int4p;
#if __has_builtin(__builtin_amdgcn_s_wait_asynccnt)
#define WAIT_ASYNC() __builtin_amdgcn_s_wait_asynccnt(0)
#else
#define WAIT_ASYNC() asm volatile("s_wait_asynccnt 0x0" ::: "memory")
#endif
#endif

// ---------------------------------------------------------------------------
// Prep: f32 weights -> f16 workspace, swizzled into WMMA B-fragment order.
// B-frag (KxN) k order used here, same as A: lane holds n = lane&15; element j -> k = (lane>>4)*16 + j.
// Fragment f = ntile*KC + kc; storage: wsH[ (f*32 + lane)*16 + j ].
// ---------------------------------------------------------------------------
__global__ void amrsrn_prep(const float* __restrict__ w0, const float* __restrict__ w1,
                            const float* __restrict__ w2, const float* __restrict__ w3,
                            const float* __restrict__ w4, const float* __restrict__ w5,
                            const float* __restrict__ w6,
                            const float* __restrict__ b0, const float* __restrict__ b1,
                            const float* __restrict__ b2, const float* __restrict__ b3,
                            const float* __restrict__ b4, const float* __restrict__ b5,
                            const float* __restrict__ b6,
                            f16* __restrict__ wsH, float* __restrict__ wsF) {
  int idx = blockIdx.x * blockDim.x + threadIdx.x;
  if (idx < W_F16_TOTAL) {
    int rem, KC, Kin;
    const float* W;
    if (idx < L0_WELEMS) { rem = idx; KC = KC0; Kin = 164; W = w0; }
    else {
      int e = idx - L0_WELEMS;
      int l = 1 + e / L1_WELEMS;
      rem = e % L1_WELEMS; KC = KC1; Kin = 256;
      W = (l == 1) ? w1 : (l == 2) ? w2 : (l == 3) ? w3 : (l == 4) ? w4 : w5;
    }
    int j     = rem & 15;
    int lane  = (rem >> 4) & 31;
    int frag  = rem >> 9;
    int ntile = frag / KC;
    int kc    = frag % KC;
    int n     = ntile * 16 + (lane & 15);
    int k     = kc * 32 + (lane >> 4) * 16 + j;
    float v = (k < Kin) ? W[n * Kin + k] : 0.0f;
    wsH[idx] = (f16)v;
  } else {
    int e = idx - W_F16_TOTAL;
    if (e < 6 * 256) {
      int l = e >> 8, i = e & 255;
      const float* B = (l == 0) ? b0 : (l == 1) ? b1 : (l == 2) ? b2
                      : (l == 3) ? b3 : (l == 4) ? b4 : b5;
      wsF[e] = B[i];
    } else if (e < 6 * 256 + 256) {
      wsF[e] = w6[e - 6 * 256];
    } else if (e == 6 * 256 + 256) {
      wsF[e] = b6[0];
    }
  }
}

// ---------------------------------------------------------------------------
// One WMMA layer: wave owns 16 rows of act. A-frags fully in regs -> in-place
// activation update is safe. Snake: 0.5x + sin(x)^2 fused on f32 accumulator.
// A-frag k order used here (16x32 MxK, same as B): lane m=lane&15, half=lane>>4; elem j ->
//   k = chunk*32 + half*16 + j
// C/D (f32 16x16): lane n=lane&15; vgpr r -> m = half*8 + r.
// ---------------------------------------------------------------------------
template <int KC>
__device__ __forceinline__ void layer_wmma(f16 (*act)[ACT_STRIDE],
                                           const f16* __restrict__ Wlds,
                                           const float* __restrict__ biasLds,
                                           int lane, int wave) {
  const int m = lane & 15, half = lane >> 4;
  const int row0 = wave * 16;
  v16h A[KC];
#pragma unroll
  for (int c = 0; c < KC; ++c) {
    const f16* ap = &act[row0 + m][c * 32 + half * 16];
    v8h lo = *(const v8h*)ap;
    v8h hi = *(const v8h*)(ap + 8);
    A[c] = __builtin_shufflevector(lo, hi, 0, 1, 2, 3, 4, 5, 6, 7,
                                           8, 9, 10, 11, 12, 13, 14, 15);
  }
#pragma unroll
  for (int nt = 0; nt < 16; ++nt) {
    v8f acc = {};
#pragma unroll
    for (int c = 0; c < KC; ++c) {
      const f16* bp = Wlds + ((size_t)(nt * KC + c) * 32 + lane) * 16;
      v8h lo = *(const v8h*)bp;
      v8h hi = *(const v8h*)(bp + 8);
      v16h B = __builtin_shufflevector(lo, hi, 0, 1, 2, 3, 4, 5, 6, 7,
                                               8, 9, 10, 11, 12, 13, 14, 15);
      acc = __builtin_amdgcn_wmma_f32_16x16x32_f16(false, A[c], false, B,
                                                   (short)0, acc, false, false);
    }
    float bv = biasLds[nt * 16 + m];
#pragma unroll
    for (int r = 0; r < 8; ++r) {
      float v = acc[r] + bv;
      float s = __sinf(v);
      v = 0.5f * v + s * s;               // SnakeAlt
      act[row0 + half * 8 + r][nt * 16 + m] = (f16)v;
    }
  }
}

// ---------------------------------------------------------------------------
// Fused kernel: transforms + trilinear gather + PE -> LDS, then 6 WMMA layers,
// then 256->1 dot product. 256 threads = 8 waves; 128 points/block.
// ---------------------------------------------------------------------------
__launch_bounds__(THREADS_C, 1)
__global__ void amrsrn_fused(const float* __restrict__ x,
                             const float* __restrict__ T,
                             const float* __restrict__ grids,
                             const f16* __restrict__ wsH,
                             const float* __restrict__ wsF,
                             float* __restrict__ out) {
  __shared__ f16   act[PPB][ACT_STRIDE];     // 66 KB
  __shared__ f16   Wlds[L1_WELEMS];          // 128 KB
  __shared__ float Tlds[32 * 16];            // 2 KB
  __shared__ float biasLds[256];
  __shared__ float w6Lds[256];

  const int t = threadIdx.x;
  const int base = blockIdx.x * PPB;

  Tlds[t]       = T[t];
  Tlds[t + 256] = T[t + 256];
  w6Lds[t]      = wsF[6 * 256 + t];
  __syncthreads();

  const int p = t & 127;
  const int P = base + p;
  const float px = x[3 * P + 0], py = x[3 * P + 1], pz = x[3 * P + 2];

  // Positional encoding + zero-pad of layer-0 K tail (threads 0..127 = waves 0-3)
  if (t < 128) {
    float c3[3] = {px, py, pz};
#pragma unroll
    for (int l = 0; l < 6; ++l) {
      float fr = (float)(1 << l) * PI_F;
#pragma unroll
      for (int d = 0; d < 3; ++d) {
        float a = c3[d] * fr;
        act[p][l * 6 + d]     = (f16)__sinf(a);
        act[p][l * 6 + 3 + d] = (f16)__cosf(a);
      }
    }
#pragma unroll
    for (int kcol = 164; kcol < 192; kcol += 4) {
      v4h zz = {};
      *(v4h*)&act[p][kcol] = zz;
    }
  }

  // Multi-grid trilinear sampling (each thread: its point, 16 of 32 grids)
  const int g0 = t >> 7;
  for (int i = 0; i < 16; ++i) {
    int g = g0 + 2 * i;
    const float* Tg = &Tlds[g * 16];
    float tx = Tg[0] * px + Tg[1] * py + Tg[2]  * pz + Tg[3];
    float ty = Tg[4] * px + Tg[5] * py + Tg[6]  * pz + Tg[7];
    float tz = Tg[8] * px + Tg[9] * py + Tg[10] * pz + Tg[11];
    float gx = (tx + 1.f) * 31.5f, gy = (ty + 1.f) * 31.5f, gz = (tz + 1.f) * 31.5f;
    float fxf = floorf(gx), fyf = floorf(gy), fzf = floorf(gz);
    int ix = (int)fxf, iy = (int)fyf, iz = (int)fzf;
    float fx = gx - fxf, fy = gy - fyf, fz = gz - fzf;
    float wx[2] = {1.f - fx, fx}, wy[2] = {1.f - fy, fy}, wz[2] = {1.f - fz, fz};
    float f0 = 0.f, f1 = 0.f, f2 = 0.f, f3 = 0.f;
    const float* bg = grids + (size_t)g * (4u * 262144u);
#pragma unroll
    for (int cz = 0; cz < 2; ++cz) {
      int z = iz + cz; bool zo = (z >= 0) && (z < 64);
      int zc = z < 0 ? 0 : (z > 63 ? 63 : z);
#pragma unroll
      for (int cy = 0; cy < 2; ++cy) {
        int yy = iy + cy; bool yo = (yy >= 0) && (yy < 64);
        int yc = yy < 0 ? 0 : (yy > 63 ? 63 : yy);
#pragma unroll
        for (int cx = 0; cx < 2; ++cx) {
          int xx = ix + cx; bool xo = (xx >= 0) && (xx < 64);
          int xc = xx < 0 ? 0 : (xx > 63 ? 63 : xx);
          float w = wz[cz] * wy[cy] * wx[cx];
          w = (zo && yo && xo) ? w : 0.f;
          size_t off = ((size_t)zc * 64 + yc) * 64 + xc;
          f0 += w * bg[off];
          f1 += w * bg[262144u + off];
          f2 += w * bg[2u * 262144u + off];
          f3 += w * bg[3u * 262144u + off];
        }
      }
    }
    v4h fv;
    fv.x = (f16)f0; fv.y = (f16)f1; fv.z = (f16)f2; fv.w = (f16)f3;
    *(v4h*)&act[p][36 + g * 4] = fv;
  }
  __syncthreads();

  // 6 snake layers via WMMA
  const f16* wptr = wsH;
  for (int l = 0; l < 6; ++l) {
    __syncthreads();   // protect Wlds reuse across layers
    const int welems = (l == 0) ? L0_WELEMS : L1_WELEMS;
    const int chunks = welems >> 3;
#ifdef ASYNC_LDS
    // CDNA5 async DMA: global -> LDS without touching VGPRs (ASYNCcnt).
    for (int i2 = t; i2 < chunks; i2 += THREADS_C) {
      __builtin_amdgcn_global_load_async_to_lds_b128(
          (as1_int4p)(f16*)(wptr + (size_t)i2 * 8),
          (as3_int4p)(Wlds + (size_t)i2 * 8),
          0, 0);
    }
    WAIT_ASYNC();
#else
    const v8h* src = (const v8h*)wptr;
    v8h* dst = (v8h*)Wlds;
    for (int i2 = t; i2 < chunks; i2 += THREADS_C) dst[i2] = src[i2];
#endif
    biasLds[t] = wsF[l * 256 + t];
    // prefetch next layer's weights while this one computes
    if (l < 5) __builtin_prefetch(wptr + welems + (size_t)t * 128, 0, 1);
    __syncthreads();
    if (l == 0) layer_wmma<KC0>(act, Wlds, biasLds, t & 31, t >> 5);
    else        layer_wmma<KC1>(act, Wlds, biasLds, t & 31, t >> 5);
    wptr += welems;
  }

  // final linear 256 -> 1: 2 threads per point, vectorized LDS reads
  __syncthreads();
  {
    const int ph = t & 127, hh = t >> 7;
    float s = 0.f;
    const f16* arow = &act[ph][hh * 128];
    const float* wv = &w6Lds[hh * 128];
#pragma unroll
    for (int i = 0; i < 128; i += 8) {
      v8h av = *(const v8h*)(arow + i);
#pragma unroll
      for (int j = 0; j < 8; ++j) s += (float)av[j] * wv[i + j];
    }
    biasLds[t] = s;   // biasLds reused as partial-sum scratch
  }
  __syncthreads();
  if (t < 128) {
    out[base + t] = biasLds[t] + biasLds[t + 128] + wsF[6 * 256 + 256];
  }
}

extern "C" void kernel_launch(void* const* d_in, const int* in_sizes, int n_in,
                              void* d_out, int out_size, void* d_ws, size_t ws_size,
                              hipStream_t stream) {
  (void)in_sizes; (void)n_in; (void)out_size; (void)ws_size;
  const float* x     = (const float*)d_in[0];
  const float* T     = (const float*)d_in[1];
  const float* grids = (const float*)d_in[2];
  const float* w[7];
  const float* b[7];
  for (int i = 0; i < 7; ++i) {
    w[i] = (const float*)d_in[3 + i];
    b[i] = (const float*)d_in[10 + i];
  }
  f16*   wsH = (f16*)d_ws;
  float* wsF = (float*)((char*)d_ws + (size_t)W_F16_TOTAL * 2);

  const int totalPrep = W_F16_TOTAL + F32_TAIL;
  amrsrn_prep<<<(totalPrep + THREADS_C - 1) / THREADS_C, THREADS_C, 0, stream>>>(
      w[0], w[1], w[2], w[3], w[4], w[5], w[6],
      b[0], b[1], b[2], b[3], b[4], b[5], b[6], wsH, wsF);

  amrsrn_fused<<<N_POINTS_C / PPB, THREADS_C, 0, stream>>>(
      x, T, grids, wsH, wsF, (float*)d_out);
}
